// CrossRegionAttentionFusion_85985245265957
// MI455X (gfx1250) — compile-verified, matrix operand order checked by reading, not run on hardware
//
#include <hip/hip_runtime.h>

// ---------------------------------------------------------------------------
// CrossRegionAttentionFusion for MI455X (gfx1250), wave32, WMMA bf16x3 path.
// Round 4: fix round-3 spills: __launch_bounds__(256,2) raises the VGPR cap
// (2 waves/SIMD is the LDS-imposed occupancy anyway), and all hot-loop
// addressing is reduced to per-stage base pointers + constant instruction
// offsets (hi->lo planes at fixed +delta, folded into the 24-bit global /
// 16-bit DS offset fields).
// ---------------------------------------------------------------------------

typedef __attribute__((ext_vector_type(16))) __bf16 v16bf;
typedef __attribute__((ext_vector_type(8)))  float  v8f;

#define TOK   16      // tokens per workgroup
#define ROWS  48      // 3 * TOK (seq-major rows: row = s*16 + t)
#define CH    256
#define C3    768
#define HEADS 8
#define HD    32
#define HW    4096    // 64*64
#define NTOK  32768   // 8*64*64

// padded LDS row strides (elements)
#define LDX   260     // f32 buffers (XF / YF / OUT)
#define LDA   264     // bf16 A-buffers, stages 0..C (528B rows: 16B aligned, bank-spread)
#define LDY   776     // bf16 QKV buffer & conv-A layout (1552B rows)

// workspace layout (elements of unsigned short)
#define N_IN   (C3 * CH)   // 196608
#define N_OUT  (CH * CH)   // 65536
#define N_CV   (CH * C3)   // 196608
#define WS_IN_HI  0
#define WS_IN_LO  (N_IN)
#define WS_OUT_HI (2 * N_IN)
#define WS_OUT_LO (2 * N_IN + N_OUT)
#define WS_CV_HI  (2 * N_IN + 2 * N_OUT)
#define WS_CV_LO  (2 * N_IN + 2 * N_OUT + N_CV)

union Frag {
    v16bf          v;
    unsigned short u[16];
    uint4          q[2];
};

__device__ __forceinline__ unsigned short f2bf(float f) {
    unsigned int u = __float_as_uint(f);
    u += 0x7FFFu + ((u >> 16) & 1u);           // round-to-nearest-even
    return (unsigned short)(u >> 16);
}
__device__ __forceinline__ float bf2f(unsigned short h) {
    return __uint_as_float(((unsigned int)h) << 16);
}

// LDS A-fragment pair (hi + lo planes) from a precomputed per-lane base.
// Lo plane sits at constant +ALO_ELEM_DELTA elements -> DS immediate offset.
#define ALO_ELEM_DELTA 12800      // (OFF_ALO - OFF_AHI) / 2
__device__ __forceinline__ void loadAp(const unsigned short* p, Frag& hi, Frag& lo) {
    hi.q[0] = *(const uint4*)(p);
    hi.q[1] = *(const uint4*)(p + 16);
    lo.q[0] = *(const uint4*)(p + ALO_ELEM_DELTA);
    lo.q[1] = *(const uint4*)(p + ALO_ELEM_DELTA + 16);
}

// Global B-fragment pair from a precomputed per-lane base; lo plane at a
// stage-constant element delta -> global instruction immediate offset.
__device__ __forceinline__ void loadBp(const unsigned short* __restrict__ p,
                                       int loDelta, Frag& hi, Frag& lo) {
    hi.q[0] = *(const uint4*)(p);
    hi.q[1] = *(const uint4*)(p + 16);
    lo.q[0] = *(const uint4*)(p + loDelta);
    lo.q[1] = *(const uint4*)(p + loDelta + 16);
}

// bf16x3 compensated MAC: acc += (Ahi+Alo)*(Bhi+Blo) (drop Alo*Blo term).
__device__ __forceinline__ v8f wmma3(v8f acc, const Frag& ah, const Frag& al,
                                     const Frag& bh, const Frag& bl) {
    acc = __builtin_amdgcn_wmma_f32_16x16x32_bf16(false, ah.v, false, bl.v,
                                                  (short)0, acc, false, false);
    acc = __builtin_amdgcn_wmma_f32_16x16x32_bf16(false, al.v, false, bh.v,
                                                  (short)0, acc, false, false);
    acc = __builtin_amdgcn_wmma_f32_16x16x32_bf16(false, ah.v, false, bh.v,
                                                  (short)0, acc, false, false);
    return acc;
}

// ---------------------------------------------------------------------------
// Weight pre-conversion: f32 -> packed bf16 hi/lo, run once per launch.
// ---------------------------------------------------------------------------
__global__ void convert_weights_kernel(const float* __restrict__ inW,
                                       const float* __restrict__ outW,
                                       const float* __restrict__ convW,
                                       unsigned short* __restrict__ ws) {
    int i = blockIdx.x * blockDim.x + threadIdx.x;
    float v;
    unsigned short *ph, *pl;
    if (i < N_IN) {
        v = inW[i];  ph = ws + WS_IN_HI + i;  pl = ws + WS_IN_LO + i;
    } else if (i < N_IN + N_OUT) {
        int j = i - N_IN;
        v = outW[j]; ph = ws + WS_OUT_HI + j; pl = ws + WS_OUT_LO + j;
    } else if (i < N_IN + N_OUT + N_CV) {
        int j = i - N_IN - N_OUT;
        v = convW[j]; ph = ws + WS_CV_HI + j; pl = ws + WS_CV_LO + j;
    } else {
        return;
    }
    unsigned short h = f2bf(v);
    *ph = h;
    *pl = f2bf(v - bf2f(h));
}

// LDS layout (dynamic, 175616 B total, lifetime-aliased):
//   XF   f32 [48][LDX]          0      .. 49920   (x, kept for residual)
//   AHI  bf16 25600 B           49920  .. 75520   (Xhi -> Ohi -> Yhi[16][LDY])
//   ALO  bf16 25600 B           75520  .. 101120  (Xlo -> Olo -> Ylo)
//   POOL 74496 B                101120 .. 175616
//        = QKV bf16 [48][LDY]                    (stage A/B)
//        = YF f32 [48][LDX] + OUT f32 [16][LDX]  (stage C/D)
#define OFF_AHI  49920
#define OFF_ALO  75520
#define OFF_POOL 101120
#define OFF_OUT  (OFF_POOL + 49920)
#define SMEM_BYTES 175616

__global__ void __launch_bounds__(256, 2) crossregion_fused_kernel(
    const float* __restrict__ cort, const float* __restrict__ subc,
    const float* __restrict__ vent,
    const float* __restrict__ in_proj_b, const float* __restrict__ out_proj_b,
    const float* __restrict__ ln_g, const float* __restrict__ ln_b,
    const float* __restrict__ conv_b,
    const unsigned short* __restrict__ wbf,
    float* __restrict__ out) {

    extern __shared__ char smem[];
    float*          XF  = (float*)smem;
    unsigned short* AHI = (unsigned short*)(smem + OFF_AHI);
    unsigned short* ALO = (unsigned short*)(smem + OFF_ALO);
    unsigned short* QKV = (unsigned short*)(smem + OFF_POOL);
    float*          YF  = (float*)(smem + OFF_POOL);
    float*          OUT = (float*)(smem + OFF_OUT);

    const int tid  = threadIdx.x;
    const int wave = tid >> 5;
    const int lane = tid & 31;

    const int n0   = blockIdx.x * TOK;     // 16 consecutive tokens, same image b
    const int bImg = n0 >> 12;
    const int hw0  = n0 & 4095;

    // per-lane fragment geometry
    const int rA  = lane & 15;             // M (A) or N (B) within tile
    const int kk0 = (lane >> 4) << 3;      // +8 K elements for the hi half-wave

    // ---- Stage 0: gather X (b,C,h,w) -> LDS [48] rows f32 + bf16 hi/lo ----
    {
        const int t  = tid & 15;           // token (contiguous in memory)
        const int cb = tid >> 4;           // channel sub-index
#pragma unroll
        for (int s = 0; s < 3; ++s) {
            const float* R = (s == 0) ? cort : ((s == 1) ? subc : vent);
            R += (size_t)bImg * (CH * HW) + hw0;
            const int row = s * TOK + t;
            for (int co = 0; co < CH; co += 16) {
                int c = co + cb;
                float val = R[(size_t)c * HW + t];   // 64B runs across 16 lanes
                XF[row * LDX + c] = val;
                unsigned short h = f2bf(val);
                AHI[row * LDA + c] = h;
                ALO[row * LDA + c] = f2bf(val - bf2f(h));
            }
        }
    }
    __syncthreads();

    // Per-lane LDS A-fragment bases for the 3 row tiles (stride LDA stages A/C)
    const unsigned short* aB0 = AHI + (0  + rA) * LDA + kk0;
    const unsigned short* aB1 = AHI + (16 + rA) * LDA + kk0;
    const unsigned short* aB2 = AHI + (32 + rA) * LDA + kk0;

    // ---- Stage A: QKV = X[48x256] @ in_proj_w^T -> LDS bf16 [48][768] ----
    // Per wave: 6 column tiles as 2 half-passes of 3; A fragments loaded once
    // per k-step and reused across the 3 column tiles (9 live accumulators).
    for (int half = 0; half < 2; ++half) {
        const int ntBase = wave * 6 + half * 3;
        const unsigned short* bB =
            wbf + WS_IN_HI + (size_t)(ntBase * 16 + rA) * CH + kk0;
        v8f acc[3][3] = {};                       // [nt][mt]
        for (int ks = 0; ks < CH; ks += 32) {
            Frag ah0, al0, ah1, al1, ah2, al2;
            loadAp(aB0 + ks, ah0, al0);
            loadAp(aB1 + ks, ah1, al1);
            loadAp(aB2 + ks, ah2, al2);
#pragma unroll
            for (int j = 0; j < 3; ++j) {
                Frag bh, bl;
                loadBp(bB + j * (16 * CH) + ks, N_IN, bh, bl);
                acc[j][0] = wmma3(acc[j][0], ah0, al0, bh, bl);
                acc[j][1] = wmma3(acc[j][1], ah1, al1, bh, bl);
                acc[j][2] = wmma3(acc[j][2], ah2, al2, bh, bl);
            }
        }
#pragma unroll
        for (int j = 0; j < 3; ++j) {
            const int   n    = (ntBase + j) * 16 + rA;
            const float bias = in_proj_b[n];
#pragma unroll
            for (int r = 0; r < 8; ++r) {
                int m = r + ((lane >> 4) << 3);
                QKV[(m)      * LDY + n] = f2bf(acc[j][0][r] + bias);
                QKV[(16 + m) * LDY + n] = f2bf(acc[j][1][r] + bias);
                QKV[(32 + m) * LDY + n] = f2bf(acc[j][2][r] + bias);
            }
        }
    }
    __syncthreads();

    // ---- Stage B: 8-head 3x3 attention per token (VALU) -> O hi/lo ----
    if (tid < TOK * HEADS) {               // 128 threads: (token, head)
        const int t = tid >> 3, h = tid & 7;
        const int rr[3] = {t, TOK + t, 2 * TOK + t};
        const float scale = 0.17677669529663687f;   // 1/sqrt(32)
        float s[3][3];
#pragma unroll
        for (int i = 0; i < 3; ++i)
#pragma unroll
            for (int j = 0; j < 3; ++j) {
                const unsigned short* qp = QKV + rr[i] * LDY + h * HD;
                const unsigned short* kp = QKV + rr[j] * LDY + CH + h * HD;
                float acc = 0.f;
                for (int d = 0; d < HD; ++d) acc += bf2f(qp[d]) * bf2f(kp[d]);
                s[i][j] = acc * scale;
            }
        const unsigned short* v0 = QKV + rr[0] * LDY + 2 * CH + h * HD;
        const unsigned short* v1 = QKV + rr[1] * LDY + 2 * CH + h * HD;
        const unsigned short* v2 = QKV + rr[2] * LDY + 2 * CH + h * HD;
#pragma unroll
        for (int i = 0; i < 3; ++i) {
            float mx = fmaxf(s[i][0], fmaxf(s[i][1], s[i][2]));
            float e0 = expf(s[i][0] - mx), e1 = expf(s[i][1] - mx), e2 = expf(s[i][2] - mx);
            float inv = 1.f / (e0 + e1 + e2);
            float a0 = e0 * inv, a1 = e1 * inv, a2 = e2 * inv;
            unsigned short* ohi = AHI + rr[i] * LDA + h * HD;
            unsigned short* olo = ALO + rr[i] * LDA + h * HD;
            for (int d = 0; d < HD; ++d) {
                float o = a0 * bf2f(v0[d]) + a1 * bf2f(v1[d]) + a2 * bf2f(v2[d]);
                unsigned short hb = f2bf(o);
                ohi[d] = hb;
                olo[d] = f2bf(o - bf2f(hb));
            }
        }
    }
    __syncthreads();

    // ---- Stage C: out-proj O[48x256] @ out_proj_w^T + bias + residual -> YF
    {
        const int ntBase = wave * 2;              // 16 col tiles / 8 waves
        const unsigned short* bB =
            wbf + WS_OUT_HI + (size_t)(ntBase * 16 + rA) * CH + kk0;
        v8f acc[2][3] = {};
        for (int ks = 0; ks < CH; ks += 32) {
            Frag ah0, al0, ah1, al1, ah2, al2;
            loadAp(aB0 + ks, ah0, al0);
            loadAp(aB1 + ks, ah1, al1);
            loadAp(aB2 + ks, ah2, al2);
#pragma unroll
            for (int j = 0; j < 2; ++j) {
                Frag bh, bl;
                loadBp(bB + j * (16 * CH) + ks, N_OUT, bh, bl);
                acc[j][0] = wmma3(acc[j][0], ah0, al0, bh, bl);
                acc[j][1] = wmma3(acc[j][1], ah1, al1, bh, bl);
                acc[j][2] = wmma3(acc[j][2], ah2, al2, bh, bl);
            }
        }
#pragma unroll
        for (int j = 0; j < 2; ++j) {
            const int   n    = (ntBase + j) * 16 + rA;
            const float bias = out_proj_b[n];
#pragma unroll
            for (int r = 0; r < 8; ++r) {
                int m = r + ((lane >> 4) << 3);
                YF[(m)      * LDX + n] = acc[j][0][r] + bias + XF[(m)      * LDX + n];
                YF[(16 + m) * LDX + n] = acc[j][1][r] + bias + XF[(16 + m) * LDX + n];
                YF[(32 + m) * LDX + n] = acc[j][2][r] + bias + XF[(32 + m) * LDX + n];
            }
        }
    }
    __syncthreads();

    // ---- LayerNorm per row, regroup to conv layout Y[16][768] bf16 hi/lo ----
    if (tid < ROWS) {
        const int row = tid;               // row = s*16 + t
        const float* y = YF + row * LDX;
        float mu = 0.f;
        for (int c = 0; c < CH; ++c) mu += y[c];
        mu *= (1.f / CH);
        float var = 0.f;
        for (int c = 0; c < CH; ++c) { float d = y[c] - mu; var += d * d; }
        var *= (1.f / CH);
        const float rstd = rsqrtf(var + 1e-5f);
        const int s = row >> 4, t = row & 15;
        unsigned short* dhi = AHI + t * LDY + s * CH;
        unsigned short* dlo = ALO + t * LDY + s * CH;
        for (int c = 0; c < CH; ++c) {
            float v = (y[c] - mu) * rstd * ln_g[c] + ln_b[c];
            unsigned short h = f2bf(v);
            dhi[c] = h;
            dlo[c] = f2bf(v - bf2f(h));
        }
    }
    __syncthreads();

    // ---- Stage D: conv1x1  Y[16x768] @ conv_w^T + bias -> OUT [16][256] ----
    {
        const int ntBase = wave * 2;
        const unsigned short* aBD = AHI + rA * LDY + kk0;
        const unsigned short* bB =
            wbf + WS_CV_HI + (size_t)(ntBase * 16 + rA) * C3 + kk0;
        v8f acc[2] = {};
        for (int ks = 0; ks < C3; ks += 32) {
            Frag ah, al;
            loadAp(aBD + ks, ah, al);
#pragma unroll
            for (int j = 0; j < 2; ++j) {
                Frag bh, bl;
                loadBp(bB + j * (16 * C3) + ks, N_CV, bh, bl);
                acc[j] = wmma3(acc[j], ah, al, bh, bl);
            }
        }
#pragma unroll
        for (int j = 0; j < 2; ++j) {
            const int   n    = (ntBase + j) * 16 + rA;
            const float bias = conv_b[n];
#pragma unroll
            for (int r = 0; r < 8; ++r) {
                int m = r + ((lane >> 4) << 3);
                OUT[m * LDX + n] = acc[j][r] + bias;
            }
        }
    }
    __syncthreads();

    // ---- Final coalesced store to (b, c, h, w) ----
    {
        const int t  = tid & 15;
        const int cb = tid >> 4;
        float* dst = out + (size_t)bImg * (CH * HW) + hw0;
        for (int co = 0; co < CH; co += 16) {
            int c = co + cb;
            dst[(size_t)c * HW + t] = OUT[t * LDX + c];
        }
    }
}

extern "C" void kernel_launch(void* const* d_in, const int* in_sizes, int n_in,
                              void* d_out, int out_size, void* d_ws, size_t ws_size,
                              hipStream_t stream) {
    (void)in_sizes; (void)n_in; (void)out_size; (void)ws_size;
    const float* cort       = (const float*)d_in[0];
    const float* subc       = (const float*)d_in[1];
    const float* vent       = (const float*)d_in[2];
    const float* in_proj_w  = (const float*)d_in[3];
    const float* in_proj_b  = (const float*)d_in[4];
    const float* out_proj_w = (const float*)d_in[5];
    const float* out_proj_b = (const float*)d_in[6];
    const float* ln_g       = (const float*)d_in[7];
    const float* ln_b       = (const float*)d_in[8];
    const float* conv_w     = (const float*)d_in[9];
    const float* conv_b     = (const float*)d_in[10];
    float* out = (float*)d_out;
    unsigned short* wbf = (unsigned short*)d_ws;

    // 1) convert weights to packed bf16 hi/lo in workspace (1.84 MB)
    convert_weights_kernel<<<(N_IN + N_OUT + N_CV + 255) / 256, 256, 0, stream>>>(
        in_proj_w, out_proj_w, conv_w, wbf);

    // 2) fused attention+LN+conv pass; opt in to >64KB dynamic LDS
    hipFuncSetAttribute((const void*)crossregion_fused_kernel,
                        hipFuncAttributeMaxDynamicSharedMemorySize, SMEM_BYTES);
    crossregion_fused_kernel<<<NTOK / TOK, 256, SMEM_BYTES, stream>>>(
        cort, subc, vent, in_proj_b, out_proj_b, ln_g, ln_b, conv_b, wbf, out);
}
